// Time_SA_52974126629181
// MI455X (gfx1250) — matrix operand order repaired and checked by reading, not compiled, not run on hardware
//
#include <hip/hip_runtime.h>
#include <math.h>

typedef __attribute__((ext_vector_type(16))) __bf16 v16bf;
typedef __attribute__((ext_vector_type(8)))  float  v8f;
typedef __attribute__((ext_vector_type(4)))  unsigned int u32x4;

#define TT    3
#define CDIM  256
#define HID   1024
#define NHEAD 8
#define DHEAD 32
#define HDIM  56
#define WDIM  56
#define HW    (HDIM*WDIM)
#define TOKPB 32          // tokens per block (kernel 1)
#define MROWS 96          // rows per block (both kernels): 32*3 = 96
#define MB_T  6           // 96/16 M tiles
#define NWAVE 8
#define LN_EPS 1e-6f

union FragBF { v16bf v; u32x4 q[2]; };

// token n, timestep t  ->  row index in the original (B,H,W) layout
__device__ __forceinline__ int map_row(int n, int t) {
    int b0  = n / HW;
    int rem = n - b0 * HW;
    int h   = rem / WDIM;
    int w   = rem - h * WDIM;
    return ((b0 * TT + t) * HDIM + h) * WDIM + w;
}

// A fragment (16x32 bf16): lane holds row (lane&15); K chunks {0..7,16..23} or
// {8..15,24..31} selected by sel = lane>>4. Two 16-byte loads, 32 bytes apart.
__device__ __forceinline__ v16bf loadA(const __bf16* base, int row, int stride, int kc, int sel) {
    FragBF f;
    const __bf16* p = base + (size_t)row * stride + kc * 32 + sel * 8;
    f.q[0] = *(const u32x4*)(p);
    f.q[1] = *(const u32x4*)(p + 16);
    return f.v;
}

// B fragment (32x16 bf16, column-major = row of W): lane holds W row (lane&15),
// K chunks {0..7,16..23} or {8..15,24..31} selected by sel. Two 16-byte loads, 32 bytes apart.
__device__ __forceinline__ v16bf loadB(const __bf16* __restrict__ base, int col, int stride, int kc, int sel) {
    FragBF f;
    const __bf16* p = base + (size_t)col * stride + kc * 32 + sel * 8;
    f.q[0] = *(const u32x4*)(p);
    f.q[1] = *(const u32x4*)(p + 16);
    return f.v;
}

__device__ __forceinline__ v8f wmma_bf16(v16bf a, v16bf b, v8f c) {
    return __builtin_amdgcn_wmma_f32_16x16x32_bf16(false, a, false, b, (short)0, c, false, false);
}

// ---------------- K0: weight f32 -> bf16 conversion ----------------
__global__ __launch_bounds__(256) void k0_cvt_bf16(
    const float* __restrict__ s0, const float* __restrict__ s1,
    const float* __restrict__ s2, const float* __restrict__ s3,
    __bf16* __restrict__ d0, __bf16* __restrict__ d1,
    __bf16* __restrict__ d2, __bf16* __restrict__ d3)
{
    int i = blockIdx.x * 256 + threadIdx.x;      // grid covers 262144
    if (i < 196608) d0[i] = (__bf16)s0[i];       // kvq_w  768x256
    if (i <  65536) d1[i] = (__bf16)s1[i];       // proj_w 256x256
    if (i < 262144) d2[i] = (__bf16)s2[i];       // fc1_w 1024x256
    if (i < 262144) d3[i] = (__bf16)s3[i];       // fc2_w  256x1024
}

// ---------------- K1: LN1 + kvq GEMM + attention(T=3) + proj + residual ----------------
__global__ __launch_bounds__(256) void k1_ln_kvq_attn_proj(
    const float* __restrict__ x,
    const float* __restrict__ n1w, const float* __restrict__ n1b,
    const __bf16* __restrict__ wkvq, const float* __restrict__ bkvq,
    const __bf16* __restrict__ wproj, const float* __restrict__ bproj,
    float* __restrict__ xt)
{
    extern __shared__ char smem[];
    __bf16* xn   = (__bf16*)(smem);                       // 96*256 bf16  = 48 KB
    __bf16* kvqo = (__bf16*)(smem + 49152);               // 96*768 bf16  = 144 KB
    __bf16* att  = (__bf16*)(smem + 49152 + 147456);      // 96*256 bf16  = 48 KB
    int*    ridx = (int*)  (smem + 49152 + 147456 + 49152);

    const int tid  = threadIdx.x;
    const int lane = tid & 31;
    const int wv   = tid >> 5;
    const int sel  = lane >> 4;
    const int ln16 = lane & 15;
    const int tokBase = blockIdx.x * TOKPB;

    if (tid < MROWS) ridx[tid] = map_row(tokBase + tid / TT, tid % TT);
    __syncthreads();

    // ---- LayerNorm 1: wave per row, lane-parallel across 256 cols ----
    for (int row = wv; row < MROWS; row += NWAVE) {
        const float* xr = x + (size_t)ridx[row] * CDIM;
        float v[8], s = 0.f, ss = 0.f;
#pragma unroll
        for (int j = 0; j < 8; ++j) { v[j] = xr[lane + 32*j]; s += v[j]; ss += v[j]*v[j]; }
#pragma unroll
        for (int o = 16; o > 0; o >>= 1) { s += __shfl_xor(s, o, 32); ss += __shfl_xor(ss, o, 32); }
        float mu = s * (1.f/CDIM);
        float rs = rsqrtf(ss * (1.f/CDIM) - mu*mu + LN_EPS);
#pragma unroll
        for (int j = 0; j < 8; ++j) {
            int c = lane + 32*j;
            xn[row*CDIM + c] = (__bf16)((v[j]-mu)*rs*n1w[c] + n1b[c]);
        }
    }
    __syncthreads();

    // ---- kvq GEMM: [96x256]x[256x768]; wave owns an N column, 6 M accumulators,
    //      each B fragment loaded once and fed to 6 back-to-back WMMAs ----
    for (int nb = wv; nb < 48; nb += NWAVE) {
        v8f acc[MB_T];
#pragma unroll
        for (int mb = 0; mb < MB_T; ++mb) acc[mb] = (v8f){0,0,0,0,0,0,0,0};
#pragma unroll
        for (int kc = 0; kc < 8; ++kc) {
            v16bf b = loadB(wkvq, nb*16 + ln16, CDIM, kc, sel);
#pragma unroll
            for (int mb = 0; mb < MB_T; ++mb) {
                v16bf a = loadA(xn, mb*16 + ln16, CDIM, kc, sel);
                acc[mb] = wmma_bf16(a, b, acc[mb]);
            }
        }
        int col = nb*16 + ln16;
        float bias = bkvq[col];
#pragma unroll
        for (int mb = 0; mb < MB_T; ++mb) {
#pragma unroll
            for (int r = 0; r < 8; ++r) {
                int m = mb*16 + sel*8 + r;
                kvqo[m*768 + col] = (__bf16)(acc[mb][r] + bias);
            }
        }
    }
    __syncthreads();

    // ---- attention: one thread per (token, head); 32 tok * 8 heads = 256 ----
    {
        const int tk = tid >> 3;
        const int hh = tid & 7;
        const int r0 = 3 * tk;
        const __bf16* kbase = kvqo +       hh*DHEAD;   // cols   0..255 = k
        const __bf16* vbase = kvqo + 256 + hh*DHEAD;   // cols 256..511 = v
        const __bf16* qbase = kvqo + 512 + hh*DHEAD;   // cols 512..767 = q
        float sc[3][3] = {{0.f,0.f,0.f},{0.f,0.f,0.f},{0.f,0.f,0.f}};
        for (int d = 0; d < DHEAD; ++d) {
            float q0 = (float)qbase[(r0+0)*768 + d];
            float q1 = (float)qbase[(r0+1)*768 + d];
            float q2 = (float)qbase[(r0+2)*768 + d];
            float k0 = (float)kbase[(r0+0)*768 + d];
            float k1 = (float)kbase[(r0+1)*768 + d];
            float k2 = (float)kbase[(r0+2)*768 + d];
            sc[0][0] += q0*k0; sc[0][1] += q0*k1; sc[0][2] += q0*k2;
            sc[1][0] += q1*k0; sc[1][1] += q1*k1; sc[1][2] += q1*k2;
            sc[2][0] += q2*k0; sc[2][1] += q2*k1; sc[2][2] += q2*k2;
        }
        const float scale = 0.17677669529663687f;   // 1/sqrt(32)
        float p[3][3];
#pragma unroll
        for (int i = 0; i < 3; ++i) {
            float a0 = sc[i][0]*scale, a1 = sc[i][1]*scale, a2 = sc[i][2]*scale;
            float mx = fmaxf(a0, fmaxf(a1, a2));
            float e0 = __expf(a0-mx), e1 = __expf(a1-mx), e2 = __expf(a2-mx);
            float inv = 1.f / (e0+e1+e2);
            p[i][0]=e0*inv; p[i][1]=e1*inv; p[i][2]=e2*inv;
        }
        for (int d = 0; d < DHEAD; ++d) {
            float v0 = (float)vbase[(r0+0)*768 + d];
            float v1 = (float)vbase[(r0+1)*768 + d];
            float v2 = (float)vbase[(r0+2)*768 + d];
#pragma unroll
            for (int i = 0; i < 3; ++i) {
                att[(r0+i)*CDIM + hh*DHEAD + d] =
                    (__bf16)(p[i][0]*v0 + p[i][1]*v1 + p[i][2]*v2);
            }
        }
    }
    __syncthreads();

    // ---- proj GEMM + bias + residual -> xt: 16 N columns / 8 waves ----
    for (int nb = wv; nb < 16; nb += NWAVE) {
        v8f acc[MB_T];
#pragma unroll
        for (int mb = 0; mb < MB_T; ++mb) acc[mb] = (v8f){0,0,0,0,0,0,0,0};
#pragma unroll
        for (int kc = 0; kc < 8; ++kc) {
            v16bf b = loadB(wproj, nb*16 + ln16, CDIM, kc, sel);
#pragma unroll
            for (int mb = 0; mb < MB_T; ++mb) {
                v16bf a = loadA(att, mb*16 + ln16, CDIM, kc, sel);
                acc[mb] = wmma_bf16(a, b, acc[mb]);
            }
        }
        int col = nb*16 + ln16;
        float bias = bproj[col];
#pragma unroll
        for (int mb = 0; mb < MB_T; ++mb) {
#pragma unroll
            for (int r = 0; r < 8; ++r) {
                int m = mb*16 + sel*8 + r;
                float val = acc[mb][r] + bias + x[(size_t)ridx[m]*CDIM + col];
                xt[((size_t)tokBase*TT + m)*CDIM + col] = val;
            }
        }
    }
}

// ---------------- K2: LN2 + fc1 + GELU + fc2 + residual + scatter store ----------------
__global__ __launch_bounds__(256) void k2_ln_mlp(
    const float* __restrict__ xt,
    const float* __restrict__ n2w, const float* __restrict__ n2b,
    const __bf16* __restrict__ wfc1, const float* __restrict__ bfc1,
    const __bf16* __restrict__ wfc2, const float* __restrict__ bfc2,
    float* __restrict__ out)
{
    extern __shared__ char smem[];
    __bf16* xn   = (__bf16*)(smem);                 // 96*256  bf16 = 48 KB
    __bf16* hbuf = (__bf16*)(smem + 49152);         // 96*1024 bf16 = 192 KB
    int*    ridx = (int*)  (smem + 49152 + 196608);

    const int tid  = threadIdx.x;
    const int lane = tid & 31;
    const int wv   = tid >> 5;
    const int sel  = lane >> 4;
    const int ln16 = lane & 15;
    const int rbase = blockIdx.x * MROWS;

    if (tid < MROWS) { int r = rbase + tid; ridx[tid] = map_row(r / TT, r % TT); }
    __syncthreads();

    // ---- LayerNorm 2 ----
    for (int row = wv; row < MROWS; row += NWAVE) {
        const float* xr = xt + (size_t)(rbase + row) * CDIM;
        float v[8], s = 0.f, ss = 0.f;
#pragma unroll
        for (int j = 0; j < 8; ++j) { v[j] = xr[lane + 32*j]; s += v[j]; ss += v[j]*v[j]; }
#pragma unroll
        for (int o = 16; o > 0; o >>= 1) { s += __shfl_xor(s, o, 32); ss += __shfl_xor(ss, o, 32); }
        float mu = s * (1.f/CDIM);
        float rs = rsqrtf(ss * (1.f/CDIM) - mu*mu + LN_EPS);
#pragma unroll
        for (int j = 0; j < 8; ++j) {
            int c = lane + 32*j;
            xn[row*CDIM + c] = (__bf16)((v[j]-mu)*rs*n2w[c] + n2b[c]);
        }
    }
    __syncthreads();

    // ---- fc1 GEMM + exact-erf GELU: 64 N columns / 8 waves, 6 M accumulators ----
    for (int nb = wv; nb < 64; nb += NWAVE) {
        v8f acc[MB_T];
#pragma unroll
        for (int mb = 0; mb < MB_T; ++mb) acc[mb] = (v8f){0,0,0,0,0,0,0,0};
#pragma unroll
        for (int kc = 0; kc < 8; ++kc) {
            v16bf b = loadB(wfc1, nb*16 + ln16, CDIM, kc, sel);
#pragma unroll
            for (int mb = 0; mb < MB_T; ++mb) {
                v16bf a = loadA(xn, mb*16 + ln16, CDIM, kc, sel);
                acc[mb] = wmma_bf16(a, b, acc[mb]);
            }
        }
        int col = nb*16 + ln16;
        float bias = bfc1[col];
#pragma unroll
        for (int mb = 0; mb < MB_T; ++mb) {
#pragma unroll
            for (int r = 0; r < 8; ++r) {
                int m = mb*16 + sel*8 + r;
                float u = acc[mb][r] + bias;
                float g = 0.5f * u * (1.f + erff(u * 0.70710678118654752f));
                hbuf[m*HID + col] = (__bf16)g;
            }
        }
    }
    __syncthreads();

    // ---- fc2 GEMM (K=1024, 32 wmma k-steps) + bias + residual -> scattered out ----
    for (int nb = wv; nb < 16; nb += NWAVE) {
        v8f acc[MB_T];
#pragma unroll
        for (int mb = 0; mb < MB_T; ++mb) acc[mb] = (v8f){0,0,0,0,0,0,0,0};
#pragma unroll 4
        for (int kc = 0; kc < 32; ++kc) {
            v16bf b = loadB(wfc2, nb*16 + ln16, HID, kc, sel);
#pragma unroll
            for (int mb = 0; mb < MB_T; ++mb) {
                v16bf a = loadA(hbuf, mb*16 + ln16, HID, kc, sel);
                acc[mb] = wmma_bf16(a, b, acc[mb]);
            }
        }
        int col = nb*16 + ln16;
        float bias = bfc2[col];
#pragma unroll
        for (int mb = 0; mb < MB_T; ++mb) {
#pragma unroll
            for (int r = 0; r < 8; ++r) {
                int m = mb*16 + sel*8 + r;
                float val = acc[mb][r] + bias + xt[(size_t)(rbase + m)*CDIM + col];
                out[(size_t)ridx[m]*CDIM + col] = val;
            }
        }
    }
}

extern "C" void kernel_launch(void* const* d_in, const int* in_sizes, int n_in,
                              void* d_out, int out_size, void* d_ws, size_t ws_size,
                              hipStream_t stream) {
    (void)in_sizes; (void)n_in; (void)out_size; (void)ws_size;
    const float* x      = (const float*)d_in[0];
    const float* n1w    = (const float*)d_in[1];
    const float* n1b    = (const float*)d_in[2];
    const float* kvq_w  = (const float*)d_in[3];
    const float* kvq_b  = (const float*)d_in[4];
    const float* proj_w = (const float*)d_in[5];
    const float* proj_b = (const float*)d_in[6];
    const float* n2w    = (const float*)d_in[7];
    const float* n2b    = (const float*)d_in[8];
    const float* fc1_w  = (const float*)d_in[9];
    const float* fc1_b  = (const float*)d_in[10];
    const float* fc2_w  = (const float*)d_in[11];
    const float* fc2_b  = (const float*)d_in[12];

    char* ws = (char*)d_ws;
    float* xt = (float*)ws;                              // 150528*256 f32
    size_t XT = (size_t)150528 * 256 * 4;
    __bf16* wkvq  = (__bf16*)(ws + XT);                  // 196608 bf16
    __bf16* wproj = wkvq + 196608;                       //  65536 bf16
    __bf16* wfc1  = wproj + 65536;                       // 262144 bf16
    __bf16* wfc2  = wfc1 + 262144;                       // 262144 bf16

    k0_cvt_bf16<<<1024, 256, 0, stream>>>(kvq_w, proj_w, fc1_w, fc2_w,
                                          wkvq, wproj, wfc1, wfc2);

    const int nBlk = 1568;                               // 50176 tok / 32 ; 150528 rows / 96
    size_t smem1 = 49152 + 147456 + 49152 + MROWS * sizeof(int);
    k1_ln_kvq_attn_proj<<<nBlk, 256, smem1, stream>>>(x, n1w, n1b, wkvq, kvq_b,
                                                      wproj, proj_b, xt);
    size_t smem2 = 49152 + 196608 + MROWS * sizeof(int);
    k2_ln_mlp<<<nBlk, 256, smem2, stream>>>(xt, n2w, n2b, wfc1, fc1_b,
                                            wfc2, fc2_b, (float*)d_out);
}
